// LSHSampledLayer_69965017252063
// MI455X (gfx1250) — matrix operand order repaired and checked by reading, not compiled, not run on hardware
//
#include <hip/hip_runtime.h>

typedef __attribute__((ext_vector_type(8)))  float  v8f;
typedef __attribute__((ext_vector_type(16))) __bf16 v16bf;
typedef __attribute__((ext_vector_type(2)))  __bf16 v2bf;
typedef __attribute__((ext_vector_type(4)))  unsigned int u32x4;
typedef __attribute__((ext_vector_type(8)))  int i32x8;
typedef __attribute__((ext_vector_type(4)))  int i32x4;

#define DEVINL __device__ __forceinline__

#if defined(__has_builtin)
#if __has_builtin(__builtin_amdgcn_tensor_load_to_lds) && __has_builtin(__builtin_amdgcn_s_wait_tensorcnt)
#define HAVE_TDM 1
#endif
#if __has_builtin(__builtin_amdgcn_cvt_pk_bf16_f32)
#define HAVE_CVT_PK_BF16 1
#endif
#endif

// ---------------- problem constants (fixed by the reference) ----------------
constexpr int N_  = 1024;
constexpr int D_  = 1024;
constexpr int C_  = 32000;
constexpr int L_  = 8;
constexpr int PJ_ = 72;              // K*L projection dims
constexpr int CAP_ = 32768;          // CAP_PAIRS
constexpr int WPR_ = C_ / 32;        // 1000 mask words per row
constexpr float ALPHA_ = 1.0f;
constexpr float EPS_   = 1e-6f;
constexpr int K1_RANK = 31967;       // int(C*(1-T1)) - 1  (0-based kth smallest)
constexpr int K2_RANK = 15999;       // int(C*(1-T2)) - 1

// ---------------- workspace layout ----------------
constexpr size_t A256(size_t x){ return (x + 255) & ~(size_t)255; }
constexpr size_t OFF_P     = 0;                                           // [N,C] f32 product
constexpr size_t OFF_QPROJ = OFF_P     + A256((size_t)N_*C_*4);           // [N,72]
constexpr size_t OFF_WPROJ = OFF_QPROJ + A256((size_t)N_*PJ_*4);          // [C,72]
constexpr size_t OFF_QFP   = OFF_WPROJ + A256((size_t)C_*PJ_*4);          // [N,8]
constexpr size_t OFF_WFP   = OFF_QFP   + A256((size_t)N_*L_*4);           // [C,8]
constexpr size_t OFF_MASK  = OFF_WFP   + A256((size_t)C_*L_*4);           // [N,1000] u32
constexpr size_t OFF_UNION = OFF_MASK  + A256((size_t)N_*WPR_*4);         // [1000] u32
constexpr size_t OFF_T1V   = OFF_UNION + A256((size_t)WPR_*4);            // [N]
constexpr size_t OFF_T2V   = OFF_T1V   + A256((size_t)N_*4);
constexpr size_t OFF_CNTP  = OFF_T2V   + A256((size_t)N_*4);
constexpr size_t OFF_CNTN  = OFF_CNTP  + A256((size_t)N_*4);
constexpr size_t OFF_BASEP = OFF_CNTN  + A256((size_t)N_*4);
constexpr size_t OFF_BASEN = OFF_BASEP + A256((size_t)N_*4);
constexpr size_t OFF_SCALF = OFF_BASEN + A256((size_t)N_*4);              // f32 scalars
constexpr size_t OFF_SCALI = OFF_SCALF + 256;                             // i32 scalars
constexpr size_t OFF_PR    = OFF_SCALI + 256;
constexpr size_t OFF_PC    = OFF_PR    + A256((size_t)CAP_*4);
constexpr size_t OFF_NR    = OFF_PC    + A256((size_t)CAP_*4);
constexpr size_t OFF_NC    = OFF_NR    + A256((size_t)CAP_*4);
constexpr size_t OFF_TLB   = OFF_NC    + A256((size_t)CAP_*4);            // [CAP] per-pair losses
constexpr size_t OFF_XB    = OFF_TLB   + A256((size_t)CAP_*4);            // [N,D] bf16 copy of x

// ---------------- helpers ----------------
DEVINL unsigned pack_bf2(float f0, float f1){      // two f32 -> packed bf16 pair (RNE)
#ifdef HAVE_CVT_PK_BF16
  v2bf r = __builtin_amdgcn_cvt_pk_bf16_f32(f0, f1);   // v_cvt_pk_bf16_f32
  return __builtin_bit_cast(unsigned, r);
#else
  unsigned a = __float_as_uint(f0);
  unsigned b = __float_as_uint(f1);
  a += 0x7FFFu + ((a >> 16) & 1u);
  b += 0x7FFFu + ((b >> 16) & 1u);
  return (a >> 16) | (b & 0xFFFF0000u);
#endif
}
DEVINL unsigned f2key(float f){                    // f32 -> order-preserving u32
  unsigned u = __float_as_uint(f);
  return (u & 0x80000000u) ? ~u : (u | 0x80000000u);
}
DEVINL float key2f(unsigned k){
  unsigned u = (k & 0x80000000u) ? (k & 0x7FFFFFFFu) : ~k;
  return __uint_as_float(u);
}
DEVINL int imin(int a, int b){ return a < b ? a : b; }

union FragBF { v16bf v; uint4 q[2]; };

// ============ x -> bf16 (one pass; x is re-read by 250 column blocks) ============
__global__ __launch_bounds__(256) void convert_x_kernel(const float* __restrict__ x,
                                                        unsigned short* __restrict__ xb){
  size_t i = ((size_t)blockIdx.x * 256 + threadIdx.x) * 4;
  float4 f = *(const float4*)(x + i);
  uint2 u;
  u.x = pack_bf2(f.x, f.y);
  u.y = pack_bf2(f.z, f.w);
  *(uint2*)(xb + i) = u;
}

// ================= GEMM: product = x @ W.T  (bf16 WMMA, f32 acc) =================
// block = 256 threads (8 waves). Block tile 128(M) x 128(N), K-step 32.
// A tile: TDM tensor_load_to_lds (one descriptor per K-step, wave0 issues,
//         TENSORcnt) with fallback to GLOBAL_LOAD_ASYNC_TO_LDS_B128 (ASYNCcnt).
// B tile: f32 W converted to bf16 (v_cvt_pk_bf16_f32) stored K-major, b32 stores.
__global__ __launch_bounds__(256) void gemm_kernel(const unsigned short* __restrict__ Xb,
                                                   const float* __restrict__ Wm,
                                                   float* __restrict__ P){
  __shared__ unsigned short As[2][128][32];   // A tile, M-major  (8 KB / buffer)
  __shared__ unsigned short Bs[2][32][128];   // W tile, K-major  (8 KB / buffer)

  const int tid  = threadIdx.x;
  const int lane = tid & 31;
  const int wave = tid >> 5;
  const int wr = wave >> 1;              // 0..3  -> M offset wr*32
  const int wc = wave & 1;               // 0..1  -> N offset wc*64
  const int r  = lane & 15;
  const int kh = lane >> 4;              // lane half
  const int koff = kh * 8;               // A frag: K chunk select
  const int kb = r + kh * 16;            // B frag: this lane's K index
  const int bm = blockIdx.y * 128;
  const int bn = blockIdx.x * 128;
  const int npair = tid >> 2;            // 0..63  -> B rows 2*npair, 2*npair+1
  const int kseg  = tid & 3;             // 0..3   -> B K chunk of 8
  const int nn = 2 * npair;
  const int kbase = kseg * 8;

  v8f acc[2][4];
  const v8f vzero = {0.f,0.f,0.f,0.f,0.f,0.f,0.f,0.f};
  #pragma unroll
  for (int a=0;a<2;++a)
    #pragma unroll
    for (int n2=0;n2<4;++n2) acc[a][n2] = vzero;

#ifdef HAVE_TDM
  // --- Tensor Data Mover: 128x32 bf16 tile of Xb -> LDS, one D# per K-step ---
  auto stageA = [&](int bsel, int k0){
    if (wave != 0) return;
    unsigned long long ga = (unsigned long long)(size_t)(Xb + (size_t)bm*D_ + k0);
    u32x4 g0; i32x8 g1; i32x4 z4;
    g0[0] = 1u;                                          // count=1, user descriptor
    g0[1] = (unsigned)(size_t)&As[bsel][0][0];           // lds_addr (bytes)
    g0[2] = (unsigned)ga;                                // global_addr[31:0]
    g0[3] = (unsigned)((ga >> 32) & 0x01FFFFFFu)         // global_addr[56:32]
          | (2u << 30);                                  // type = 2 ("image")
    g1[0] = (int)(1u << 16);                             // data_size = 1 -> 2 bytes
    g1[1] = 0;                                           // tensor_dim0 = 1<<20 (low16=0)
    g1[2] = (int)0x10;                                   // dim0[31:16]=0x10 | dim1 low16=0
    g1[3] = (int)(0x10u | (32u << 16));                  // dim1[31:16]=0x10 | tile_dim0=32
    g1[4] = 128;                                         // tile_dim1 = 128 rows
    g1[5] = (int)D_;                                     // tensor_dim0_stride = D elems
    g1[6] = 0;
    g1[7] = 0;
    z4[0] = z4[1] = z4[2] = z4[3] = 0;                   // groups 2/3 unused (<=2D tile)
#if defined(__clang_major__) && (__clang_major__ >= 23)
    i32x8 z8;
    #pragma unroll
    for (int q=0;q<8;++q) z8[q] = 0;
    __builtin_amdgcn_tensor_load_to_lds(g0, g1, z4, z4, z8, 0);
#else
    __builtin_amdgcn_tensor_load_to_lds(g0, g1, z4, z4, 0);
#endif
  };
  auto waitA = [&](){
    if (wave == 0) __builtin_amdgcn_s_wait_tensorcnt(0);
  };
#else
  // --- fallback: per-lane async DMA of A tile (bf16) straight into LDS ---
  auto stageA = [&](int bsel, int k0){
    #pragma unroll
    for (int t=0;t<2;++t){
      int idx = tid + t*256;             // 512 x 16B chunks
      int row = idx >> 2, seg = idx & 3;
      const unsigned short* gp = Xb + (size_t)(bm+row)*D_ + k0 + seg*8;
      unsigned lofs = (unsigned)(size_t)&As[bsel][row][seg*8];
      asm volatile("global_load_async_to_lds_b128 %0, %1, off"
                   :: "v"(lofs), "v"(gp) : "memory");
    }
  };
  auto waitA = [&](){
    asm volatile("s_wait_asynccnt 0x0" ::: "memory");
  };
#endif

  // --- B tile: 2 N-rows x 8 K per thread; pack bf16 pairs along N ---
  float4 A0, A1, B0, B1;
  auto fetchB = [&](int k0){
    const float* r0 = Wm + (size_t)(bn + nn)*D_ + k0 + kbase;
    A0 = *(const float4*)r0;        A1 = *(const float4*)(r0 + 4);
    B0 = *(const float4*)(r0 + D_); B1 = *(const float4*)(r0 + D_ + 4);
  };
  auto stashB = [&](int bsel){
    *(unsigned*)&Bs[bsel][kbase+0][nn] = pack_bf2(A0.x, B0.x);
    *(unsigned*)&Bs[bsel][kbase+1][nn] = pack_bf2(A0.y, B0.y);
    *(unsigned*)&Bs[bsel][kbase+2][nn] = pack_bf2(A0.z, B0.z);
    *(unsigned*)&Bs[bsel][kbase+3][nn] = pack_bf2(A0.w, B0.w);
    *(unsigned*)&Bs[bsel][kbase+4][nn] = pack_bf2(A1.x, B1.x);
    *(unsigned*)&Bs[bsel][kbase+5][nn] = pack_bf2(A1.y, B1.y);
    *(unsigned*)&Bs[bsel][kbase+6][nn] = pack_bf2(A1.z, B1.z);
    *(unsigned*)&Bs[bsel][kbase+7][nn] = pack_bf2(A1.w, B1.w);
  };

  stageA(0, 0);
  fetchB(0);
  stashB(0);
  waitA();
  __syncthreads();

  int buf = 0;
  #pragma unroll 1
  for (int ks = 0; ks < D_/32; ++ks){
    const bool more = (ks + 1 < D_/32);
    if (more){
      stageA(buf ^ 1, (ks+1)*32);        // DMA next A tile while computing
      fetchB((ks+1)*32);
    }
    if (ks + 2 < D_/32)                  // L2 prefetch two steps ahead
      __builtin_prefetch(Wm + (size_t)(bn + nn)*D_ + (ks+2)*32, 0, 1);

    FragBF fa[2], fb[4];
    #pragma unroll
    for (int mt=0; mt<2; ++mt){
      const unsigned short* ap = &As[buf][wr*32 + mt*16 + r][koff];
      fa[mt].q[0] = *(const uint4*)ap;          // K = koff..koff+7
      fa[mt].q[1] = *(const uint4*)(ap + 16);   // K = 16+koff..16+koff+7
    }
    #pragma unroll
    for (int nt=0; nt<4; ++nt){
      const unsigned short* bp = &Bs[buf][0][wc*64 + nt*16 + r];
      #pragma unroll
      for (int s=0; s<16; ++s)                  // N = r; K = koff+s (s<8), 16+koff+s-8 (s>=8)
        fb[nt].v[s] = __builtin_bit_cast(__bf16, bp[((s < 8) ? (koff + s) : (16 + koff + s - 8)) * 128]);
    }
    #pragma unroll
    for (int mt=0; mt<2; ++mt)
      #pragma unroll
      for (int nt=0; nt<4; ++nt)
        acc[mt][nt] = __builtin_amdgcn_wmma_f32_16x16x32_bf16(
            false, fa[mt].v, false, fb[nt].v, (short)0, acc[mt][nt], false, false);

    if (more){
      stashB(buf ^ 1);
      waitA();
      __syncthreads();
      buf ^= 1;
    }
  }

  // D layout: lane half selects M+8, element e adds to M; N = lane&15
  #pragma unroll
  for (int mt=0; mt<2; ++mt){
    #pragma unroll
    for (int nt=0; nt<4; ++nt){
      int gm = bm + wr*32 + mt*16 + kh*8;
      int gn = bn + wc*64 + nt*16 + r;
      float* po = P + (size_t)gm*C_ + gn;
      #pragma unroll
      for (int e=0; e<8; ++e)
        po[(size_t)e*C_] = acc[mt][nt][e];
    }
  }
}

// ============ projections (query@rp, wcat@rp) + SimHash fingerprints ============
__global__ __launch_bounds__(256) void proj_fp_kernel(const float* __restrict__ x,
    const float* __restrict__ Wm, const float* __restrict__ bia,
    const float* __restrict__ rp, float* __restrict__ qproj, float* __restrict__ wproj,
    int* __restrict__ qfp, int* __restrict__ wfp){
  const int wave = threadIdx.x >> 5, lane = threadIdx.x & 31;
  const int row = blockIdx.x * 8 + wave;
  if (row >= N_ + C_) return;
  const float* v; float last; float* proj; int* fpo;
  if (row < N_){ v = x + (size_t)row*D_; last = 1.0f;
                 proj = qproj + (size_t)row*PJ_; fpo = qfp + row*L_; }
  else { int c = row - N_; v = Wm + (size_t)c*D_; last = bia[c];
         proj = wproj + (size_t)c*PJ_; fpo = wfp + c*L_; }
  int cur = 0;
  for (int j = 0; j < PJ_; ++j){
    float s = 0.f;
    for (int d = lane; d < D_; d += 32) s += v[d] * rp[(size_t)d*PJ_ + j];
    if (lane == 0) s += last * rp[(size_t)D_*PJ_ + j];
    #pragma unroll
    for (int off = 16; off; off >>= 1) s += __shfl_down(s, off, 32);
    if (lane == 0){
      proj[j] = s;
      int k = j % 9;
      if (k == 0) cur = 0;
      cur |= (int)(s > 0.f) << k;
      if (k == 8) fpo[j / 9] = cur;
    }
  }
}

// ============ retrieval mask ============
__global__ __launch_bounds__(256) void mask_kernel(const int* __restrict__ qfp,
    const int* __restrict__ wfp, unsigned* __restrict__ mask){
  const int wave = threadIdx.x >> 5, lane = threadIdx.x & 31;
  const int wg = blockIdx.x * 8 + wave;
  const int i = wg / WPR_, word = wg % WPR_;
  const int c = word * 32 + lane;
  bool r = false;
  #pragma unroll
  for (int l = 0; l < L_; ++l) r |= (qfp[i*L_ + l] == wfp[c*L_ + l]);
  unsigned bal = (unsigned)__ballot(r);
  if (lane == 0) mask[(size_t)i*WPR_ + word] = bal;
}

__global__ __launch_bounds__(256) void union_kernel(const unsigned* __restrict__ mask,
                                                    unsigned* __restrict__ uw){
  int word = blockIdx.x * 256 + threadIdx.x;
  if (word >= WPR_) return;
  unsigned u = 0;
  for (int i = 0; i < N_; ++i) u |= mask[(size_t)i*WPR_ + word];
  uw[word] = u;
}

// ============ ordered compaction of union -> sample_ids ============
__global__ __launch_bounds__(1024) void sample_kernel(const unsigned* __restrict__ uw,
                                                      float* __restrict__ ids_f){
  __shared__ unsigned cnt[WPR_];
  const int t = threadIdx.x;
  for (int s = t; s < C_; s += 1024) ids_f[s] = 0.0f;
  if (t < WPR_) cnt[t] = (unsigned)__popc(uw[t]);
  __syncthreads();
  if (t == 0){
    unsigned run = 0;
    for (int w = 0; w < WPR_; ++w){ unsigned c = cnt[w]; cnt[w] = run; run += c; }
  }
  __syncthreads();
  if (t < WPR_){
    unsigned w = uw[t]; unsigned base = cnt[t];
    while (w){
      int bit = __ffs((int)w) - 1;
      if (base < (unsigned)C_) ids_f[base] = (float)(t*32 + bit);
      ++base;
      w &= w - 1u;
    }
  }
}

// ============ exact per-row kth-smallest (two ranks) via 4-pass radix select ============
__global__ __launch_bounds__(256) void select_kernel(const float* __restrict__ P,
                                                     float* __restrict__ t1v,
                                                     float* __restrict__ t2v){
  __shared__ unsigned hist[256];
  __shared__ unsigned sh_prefix;
  __shared__ int sh_rem;
  const int row = blockIdx.x, tid = threadIdx.x;
  const float* prow = P + (size_t)row*C_;
  for (int sel = 0; sel < 2; ++sel){
    if (tid == 0){ sh_prefix = 0; sh_rem = sel ? K2_RANK : K1_RANK; }
    __syncthreads();
    for (int pass = 0; pass < 4; ++pass){
      hist[tid] = 0;
      __syncthreads();
      const unsigned pfx = sh_prefix;
      const int shift = 24 - pass*8;
      for (int e = tid; e < C_; e += 256){
        unsigned u = f2key(prow[e]);
        if (pass == 0 || (u >> (shift + 8)) == pfx)
          atomicAdd(&hist[(u >> shift) & 255u], 1u);
      }
      __syncthreads();
      if (tid == 0){
        int rem = sh_rem; unsigned b = 0;
        for (; b < 256; ++b){ unsigned h = hist[b]; if ((int)h > rem) break; rem -= (int)h; }
        sh_prefix = (pfx << 8) | b;
        sh_rem = rem;
      }
      __syncthreads();
    }
    if (tid == 0) (sel ? t2v : t1v)[row] = key2f(sh_prefix);
    __syncthreads();
  }
}

__global__ __launch_bounds__(1024) void mean_kernel(const float* __restrict__ t1v,
    const float* __restrict__ t2v, float* __restrict__ scalF){
  __shared__ float s1[1024], s2[1024];
  const int t = threadIdx.x;
  s1[t] = t1v[t]; s2[t] = t2v[t];
  __syncthreads();
  for (int o = 512; o; o >>= 1){ if (t < o){ s1[t] += s1[t+o]; s2[t] += s2[t+o]; } __syncthreads(); }
  if (t == 0){ scalF[0] = fmaxf(0.f, s1[0] / (float)N_); scalF[1] = s2[0] / (float)N_; }
}

// ============ pair counting / ordered pair extraction ============
__global__ __launch_bounds__(256) void pair_count_kernel(const float* __restrict__ P,
    const unsigned* __restrict__ mask, const float* __restrict__ scalF,
    int* __restrict__ cntp, int* __restrict__ cntn){
  const int wave = threadIdx.x >> 5, lane = threadIdx.x & 31;
  const int i = blockIdx.x * 8 + wave;
  const float t1 = scalF[0], t2 = scalF[1];
  const float* prow = P + (size_t)i*C_;
  const unsigned* mrow = mask + (size_t)i*WPR_;
  int cp = 0, cn = 0;
  for (int w = 0; w < WPR_; ++w){
    float p = prow[w*32 + lane];
    unsigned mb = (mrow[w] >> lane) & 1u;
    cp += __popc((unsigned)__ballot((p > t1) && !mb));
    cn += __popc((unsigned)__ballot((p < t2) &&  mb));
  }
  if (lane == 0){ cntp[i] = cp; cntn[i] = cn; }
}

__global__ void scan_kernel(const int* cntp, const int* cntn,
                            int* basep, int* basen, int* scalI){
  int rp_ = 0, rn = 0;
  for (int i = 0; i < N_; ++i){
    basep[i] = rp_; rp_ += cntp[i];
    basen[i] = rn;  rn  += cntn[i];
  }
  scalI[0] = rp_; scalI[1] = rn;
}

__global__ __launch_bounds__(256) void pair_write_kernel(const float* __restrict__ P,
    const unsigned* __restrict__ mask, const float* __restrict__ scalF,
    const int* __restrict__ basep, const int* __restrict__ basen,
    int* __restrict__ pr, int* __restrict__ pc, int* __restrict__ nr, int* __restrict__ nc){
  const int wave = threadIdx.x >> 5, lane = threadIdx.x & 31;
  const int i = blockIdx.x * 8 + wave;
  const float t1 = scalF[0], t2 = scalF[1];
  const float* prow = P + (size_t)i*C_;
  const unsigned* mrow = mask + (size_t)i*WPR_;
  int ip = basep[i], in_ = basen[i];
  const unsigned pre_mask = (lane == 0) ? 0u : (0xFFFFFFFFu >> (32 - lane));
  for (int w = 0; w < WPR_; ++w){
    int c = w*32 + lane;
    float p = prow[c];
    unsigned mb = (mrow[w] >> lane) & 1u;
    bool pos = (p > t1) && !mb;
    bool neg = (p < t2) &&  mb;
    unsigned balp = (unsigned)__ballot(pos);
    unsigned baln = (unsigned)__ballot(neg);
    int gp = ip  + __popc(balp & pre_mask);
    int gn = in_ + __popc(baln & pre_mask);
    if (pos && gp < CAP_){ pr[gp] = i; pc[gp] = c; }
    if (neg && gn < CAP_){ nr[gn] = i; nc[gn] = c; }
    ip  += __popc(balp);
    in_ += __popc(baln);
  }
}

// ============ triplet loss over cached 72-d projections ============
DEVINL float cosine72(const float* __restrict__ a, const float* __restrict__ b){
  float ab = 0.f, aa = 0.f, bb = 0.f;
  #pragma unroll 8
  for (int j = 0; j < PJ_; ++j){ float x = a[j], y = b[j]; ab += x*y; aa += x*x; bb += y*y; }
  float na = fmaxf(sqrtf(aa), EPS_), nb = fmaxf(sqrtf(bb), EPS_);
  return ab / (na * nb);
}

__global__ __launch_bounds__(256) void triplet_kernel(const float* __restrict__ qproj,
    const float* __restrict__ wproj, const int* __restrict__ pr, const int* __restrict__ pc,
    const int* __restrict__ nr, const int* __restrict__ nc, const int* __restrict__ scalI,
    float* __restrict__ tlb){
  const int i = blockIdx.x * 256 + threadIdx.x;
  const int m = imin(imin(scalI[0], scalI[1]), CAP_);
  float tl = 0.f;
  if (i < m){
    float sp = cosine72(qproj + (size_t)pr[i]*PJ_, wproj + (size_t)pc[i]*PJ_);
    float sn = cosine72(qproj + (size_t)nr[i]*PJ_, wproj + (size_t)nc[i]*PJ_);
    tl = fmaxf(sn - sp + ALPHA_, 0.f);
  }
  tlb[i] = tl;
}

__global__ __launch_bounds__(1024) void tl_reduce_kernel(const float* __restrict__ tlb,
                                                         float* __restrict__ scalF){
  __shared__ float sh[1024];
  const int t = threadIdx.x;
  float s = 0.f;
  for (int i = t; i < CAP_; i += 1024) s += tlb[i];
  sh[t] = s;
  __syncthreads();
  for (int o = 512; o; o >>= 1){ if (t < o) sh[t] += sh[t+o]; __syncthreads(); }
  if (t == 0) scalF[2] = sh[0];
}

__global__ void finalize_kernel(const float* scalF, const int* scalI, float* out){
  int m = imin(imin(scalI[0], scalI[1]), CAP_);
  float v = (m > 0) ? (scalF[2] / (float)m) : 0.0f;
  if (v != v) v = 0.0f;                      // nan_to_num
  out[(size_t)N_*C_ + C_] = v;
}

// ============ sampled logits = gather from product (+ bias) ============
__global__ __launch_bounds__(256) void gather_kernel(const float* __restrict__ P,
    const float* __restrict__ bia, const float* __restrict__ ids_f, float* __restrict__ out){
  size_t idx = (size_t)blockIdx.x * 256 + threadIdx.x;   // over N*C
  int s = (int)(idx % C_);
  size_t i = idx / C_;
  int id = (int)ids_f[s];
  out[idx] = P[i*C_ + id] + bia[id];
}

__global__ void init_kernel(float* scalF, int* scalI){
  int t = threadIdx.x;
  if (t < 8){ scalF[t] = 0.f; scalI[t] = 0; }
}

// =============================== host ===============================
extern "C" void kernel_launch(void* const* d_in, const int* in_sizes, int n_in,
                              void* d_out, int out_size, void* d_ws, size_t ws_size,
                              hipStream_t stream){
  (void)in_sizes; (void)n_in; (void)out_size; (void)ws_size;
  const float* x  = (const float*)d_in[0];
  const float* W  = (const float*)d_in[1];
  const float* b  = (const float*)d_in[2];
  const float* rp = (const float*)d_in[3];
  float* out = (float*)d_out;
  char* ws = (char*)d_ws;

  float*          P     = (float*)   (ws + OFF_P);
  float*          qproj = (float*)   (ws + OFF_QPROJ);
  float*          wproj = (float*)   (ws + OFF_WPROJ);
  int*            qfp   = (int*)     (ws + OFF_QFP);
  int*            wfp   = (int*)     (ws + OFF_WFP);
  unsigned*       mask  = (unsigned*)(ws + OFF_MASK);
  unsigned*       uw    = (unsigned*)(ws + OFF_UNION);
  float*          t1v   = (float*)   (ws + OFF_T1V);
  float*          t2v   = (float*)   (ws + OFF_T2V);
  int*            cntp  = (int*)     (ws + OFF_CNTP);
  int*            cntn  = (int*)     (ws + OFF_CNTN);
  int*            basep = (int*)     (ws + OFF_BASEP);
  int*            basen = (int*)     (ws + OFF_BASEN);
  float*          scalF = (float*)   (ws + OFF_SCALF);
  int*            scalI = (int*)     (ws + OFF_SCALI);
  int*            pr    = (int*)     (ws + OFF_PR);
  int*            pc    = (int*)     (ws + OFF_PC);
  int*            nr    = (int*)     (ws + OFF_NR);
  int*            nc    = (int*)     (ws + OFF_NC);
  float*          tlb   = (float*)   (ws + OFF_TLB);
  unsigned short* xb    = (unsigned short*)(ws + OFF_XB);

  float* ids_f = out + (size_t)N_*C_;   // sample_ids region of d_out

  init_kernel<<<1, 64, 0, stream>>>(scalF, scalI);
  convert_x_kernel<<<(N_*D_)/1024, 256, 0, stream>>>(x, xb);
  proj_fp_kernel<<<(N_ + C_) / 8, 256, 0, stream>>>(x, W, b, rp, qproj, wproj, qfp, wfp);
  gemm_kernel<<<dim3(C_/128, N_/128), 256, 0, stream>>>(xb, W, P);
  mask_kernel<<<(N_ * WPR_) / 8, 256, 0, stream>>>(qfp, wfp, mask);
  union_kernel<<<(WPR_ + 255) / 256, 256, 0, stream>>>(mask, uw);
  sample_kernel<<<1, 1024, 0, stream>>>(uw, ids_f);
  select_kernel<<<N_, 256, 0, stream>>>(P, t1v, t2v);
  mean_kernel<<<1, 1024, 0, stream>>>(t1v, t2v, scalF);
  pair_count_kernel<<<N_/8, 256, 0, stream>>>(P, mask, scalF, cntp, cntn);
  scan_kernel<<<1, 1, 0, stream>>>(cntp, cntn, basep, basen, scalI);
  pair_write_kernel<<<N_/8, 256, 0, stream>>>(P, mask, scalF, basep, basen, pr, pc, nr, nc);
  triplet_kernel<<<CAP_/256, 256, 0, stream>>>(qproj, wproj, pr, pc, nr, nc, scalI, tlb);
  tl_reduce_kernel<<<1, 1024, 0, stream>>>(tlb, scalF);
  finalize_kernel<<<1, 1, 0, stream>>>(scalF, scalI, out);
  gather_kernel<<<((size_t)N_*C_) / 256, 256, 0, stream>>>(P, b, ids_f, out);
}
